// SpanNER_81905026335233
// MI455X (gfx1250) — matrix operand order repaired and checked by reading, not compiled, not run on hardware
//
#include <hip/hip_runtime.h>
#include <hip/hip_bf16.h>

typedef __attribute__((ext_vector_type(16))) _Float16 v16h;
typedef __attribute__((ext_vector_type(8)))  float    v8f;

// Problem constants
constexpr int S      = 256;
constexpr int D      = 768;
constexpr int PD     = 128;
constexpr int NVALID = 765;              // 254*3 + 2 + 1
constexpr int TOTAL  = 32896;            // S*(S+1)/2
constexpr int MROWS  = 1536;             // 2*765 = 1530 padded to 16
constexpr int MT     = MROWS / 16;       // 96 row tiles
constexpr int K1     = 3 * D;            // 2304
constexpr int KT1    = K1 / 32;          // 72
constexpr int NT1    = D / 16;           // 48
constexpr int KT2    = D / 32;           // 24
constexpr int NT3    = PD / 16;          // 8

// ---------------------------------------------------------------------------
// Pad constants: logit_pad = b_span.w_cls + b_cls ;
// proj_pad = relu(b_span @ w_p1 + b_p1) @ w_p2 + b_p2
// ---------------------------------------------------------------------------
__global__ void consts_kernel(const float* __restrict__ b_span,
                              const float* __restrict__ w_p1,
                              const float* __restrict__ b_p1,
                              const float* __restrict__ w_p2,
                              const float* __restrict__ b_p2,
                              const float* __restrict__ w_cls,
                              const float* __restrict__ b_cls,
                              float* __restrict__ proj_pad,
                              float* __restrict__ logit_pad) {
  __shared__ float t[D];
  int tid = threadIdx.x;                 // 256 threads
  for (int k = tid; k < D; k += 256) {
    float acc = b_p1[k];
    for (int d = 0; d < D; ++d) acc += b_span[d] * w_p1[(size_t)d * D + k];
    t[k] = fmaxf(acc, 0.f);
  }
  __syncthreads();
  if (tid < PD) {
    float acc = b_p2[tid];
    for (int k = 0; k < D; ++k) acc += t[k] * w_p2[(size_t)k * PD + tid];
    proj_pad[tid] = acc;
  }
  if (tid == 255) {
    float acc = b_cls[0];
    for (int d = 0; d < D; ++d) acc += b_span[d] * w_cls[d];
    *logit_pad = acc;
  }
}

// ---------------------------------------------------------------------------
// Pack a row-major f32 weight (K x N) into f16 B-fragment layout:
// frag f = kt*Nt + nt ; per lane: n = nt*16 + lane%16, K = kt*32+(lane/16)*8+e (e<8), kt*32+16+(lane/16)*8+(e-8) (e>=8)
// ---------------------------------------------------------------------------
__global__ void prep_weights(const float* __restrict__ w, _Float16* __restrict__ dst,
                             int N, int Nt) {
  int f    = blockIdx.x;
  int kt   = f / Nt;
  int nt   = f % Nt;
  int lane = threadIdx.x;
  int n     = nt * 16 + (lane & 15);
  int kbase = kt * 32 + (lane >> 4) * 8;
  _Float16* o = dst + ((size_t)f * 32 + lane) * 16;
#pragma unroll
  for (int e = 0; e < 16; ++e)
    o[e] = (_Float16)w[(size_t)(kbase + e + (e >= 8 ? 8 : 0)) * N + n];
}

// ---------------------------------------------------------------------------
// Gather span rows -> f16 A-fragment layout.
// Row g (< 1530): b = g/765, v = g%765 ; v -> (i, off), j = i + off.
// A row = [hidden[b,i,:] | hidden[b,j,:] | width_emb[off,:]]
// ---------------------------------------------------------------------------
__global__ void prep_A(const float* __restrict__ hidden,
                       const float* __restrict__ width_emb,
                       _Float16* __restrict__ dst) {
  int f    = blockIdx.x;                 // rt*KT1 + kt
  int rt   = f / KT1;
  int kt   = f % KT1;
  int lane = threadIdx.x;
  int m     = lane & 15;
  int khalf = lane >> 4;
  int g     = rt * 16 + m;
  _Float16* o = dst + ((size_t)f * 32 + lane) * 16;
  if (g >= 2 * NVALID) {
#pragma unroll
    for (int e = 0; e < 16; ++e) o[e] = (_Float16)0.f;
    return;
  }
  int b = g / NVALID, v = g % NVALID;
  int i, off;
  if (v < 762)      { i = v / 3; off = v % 3; }
  else if (v < 764) { i = 254;   off = v - 762; }
  else              { i = 255;   off = 0; }
  int j = i + off;
  const float* hb = hidden + (size_t)b * S * D;
#pragma unroll
  for (int e = 0; e < 16; ++e) {
    int kloc = (e < 8) ? (khalf * 8 + e) : (16 + khalf * 8 + (e - 8));
    int k = kt * 32 + kloc;
    float x;
    if (k < D)            x = hb[(size_t)i * D + k];
    else if (k < 2 * D)   x = hb[(size_t)j * D + (k - D)];
    else                  x = width_emb[(size_t)off * D + (k - 2 * D)];
    o[e] = (_Float16)x;
  }
}

// ---------------------------------------------------------------------------
// WMMA GEMM: one wave (32 threads) per 16x16 C tile; K-loop of v_wmma.
// MODE 0: C -> f32 row-major (span) + f16 A-packed (span for GEMM2)
// MODE 1: relu(C) -> f16 A-packed (h for GEMM3)
// MODE 2: C -> proj region of d_out with (b, span) row mapping
// ---------------------------------------------------------------------------
template <int MODE>
__global__ void gemm_kernel(const _Float16* __restrict__ A,
                            const _Float16* __restrict__ Bp,
                            const float*    __restrict__ bias,
                            float*          __restrict__ Cf32,
                            _Float16*       __restrict__ Cpacked,
                            float*          __restrict__ outProj,
                            int Ktiles, int Ntiles, int KtOut) {
  int f    = blockIdx.x;
  int rt   = f / Ntiles;
  int nt   = f % Ntiles;
  int lane = threadIdx.x;

  v8f c = {};
  const v16h* Af = (const v16h*)A;
  const v16h* Bf = (const v16h*)Bp;
  for (int kt = 0; kt < Ktiles; ++kt) {
    v16h a = Af[((size_t)rt * Ktiles + kt) * 32 + lane];
    v16h b = Bf[((size_t)kt * Ntiles + nt) * 32 + lane];
    c = __builtin_amdgcn_wmma_f32_16x16x32_f16(false, a, false, b,
                                               (short)0, c, false, false);
  }

  int Ncol  = nt * 16 + (lane & 15);
  int khalf = lane >> 4;
  float bv  = bias[Ncol];
#pragma unroll
  for (int v = 0; v < 8; ++v) {
    float val = c[v] + bv;
    if (MODE == 1) val = fmaxf(val, 0.f);
    int M   = v + 8 * khalf;
    int row = rt * 16 + M;
    if (MODE == 0) {
      Cf32[(size_t)row * D + Ncol] = val;
    }
    if (MODE == 0 || MODE == 1) {
      // scatter into A-fragment layout for the next GEMM (K' = Ncol)
      int kt2 = Ncol >> 5;
      int kin = Ncol & 31;
      int e, kh2;
      if (kin < 16) { kh2 = kin >> 3; e = kin & 7; }
      else          { kh2 = (kin - 16) >> 3; e = 8 + ((kin - 16) & 7); }
      int lane2 = kh2 * 16 + M;
      Cpacked[(((size_t)rt * KtOut + kt2) * 32 + lane2) * 16 + e] = (_Float16)val;
    }
    if (MODE == 2) {
      if (row < 2 * NVALID) {
        int bI = row / NVALID, sv = row % NVALID;
        outProj[((size_t)bI * TOTAL + sv) * PD + Ncol] = val;
      }
    }
  }
}

// ---------------------------------------------------------------------------
__global__ void logits_kernel(const float* __restrict__ span,
                              const float* __restrict__ w_cls,
                              const float* __restrict__ b_cls,
                              float* __restrict__ out) {
  int t = blockIdx.x * blockDim.x + threadIdx.x;
  if (t >= 2 * NVALID) return;
  int b = t / NVALID, v = t % NVALID;
  const float* row = span + (size_t)t * D;
  float acc = b_cls[0];
  for (int k = 0; k < D; ++k) acc += row[k] * w_cls[k];
  out[(size_t)b * TOTAL + v] = acc;
}

__global__ void fillpad_kernel(const float* __restrict__ proj_pad,
                               const float* __restrict__ logit_pad,
                               float* __restrict__ out) {
  const size_t NPAD  = TOTAL - NVALID;           // 32131
  const size_t npadL = 2 * NPAD;
  const size_t npadP = 2 * NPAD * (size_t)PD;
  const size_t total = npadL + npadP;
  float lp = *logit_pad;
  for (size_t idx = (size_t)blockIdx.x * blockDim.x + threadIdx.x;
       idx < total; idx += (size_t)gridDim.x * blockDim.x) {
    if (idx < npadL) {
      size_t b = idx / NPAD, v = idx % NPAD + NVALID;
      out[b * TOTAL + v] = lp;
    } else {
      size_t r  = idx - npadL;
      size_t n  = r % PD;
      size_t rv = r / PD;
      size_t b  = rv / NPAD, v = rv % NPAD + NVALID;
      out[2 * (size_t)TOTAL + (b * TOTAL + v) * PD + n] = proj_pad[n];
    }
  }
}

// ---------------------------------------------------------------------------
extern "C" void kernel_launch(void* const* d_in, const int* in_sizes, int n_in,
                              void* d_out, int out_size, void* d_ws, size_t ws_size,
                              hipStream_t stream) {
  const float* hidden    = (const float*)d_in[0];
  const float* width_emb = (const float*)d_in[1];
  const float* w_span    = (const float*)d_in[2];
  const float* b_span    = (const float*)d_in[3];
  const float* w_cls     = (const float*)d_in[4];
  const float* b_cls     = (const float*)d_in[5];
  const float* w_p1      = (const float*)d_in[6];
  const float* b_p1      = (const float*)d_in[7];
  const float* w_p2      = (const float*)d_in[8];
  const float* b_p2      = (const float*)d_in[9];
  float* out = (float*)d_out;

  // Workspace carve-up (256B aligned chunks)
  char* base = (char*)d_ws;
  size_t off = 0;
  auto carve = [&](size_t bytes) {
    size_t o = off;
    off += (bytes + 255) & ~(size_t)255;
    return o;
  };
  _Float16* wspan_p = (_Float16*)(base + carve((size_t)K1 * D * 2));   // 3.5 MB
  _Float16* wp1_p   = (_Float16*)(base + carve((size_t)D * D * 2));    // 1.2 MB
  _Float16* wp2_p   = (_Float16*)(base + carve((size_t)D * PD * 2));   // 0.2 MB
  _Float16* a1_p    = (_Float16*)(base + carve((size_t)MROWS * K1 * 2)); // 7.1 MB
  float*    span32  = (float*)   (base + carve((size_t)MROWS * D * 4)); // 4.7 MB
  _Float16* span_p  = (_Float16*)(base + carve((size_t)MROWS * D * 2)); // 2.4 MB
  _Float16* h_p     = (_Float16*)(base + carve((size_t)MROWS * D * 2)); // 2.4 MB
  float*    projpad = (float*)   (base + carve(PD * 4));
  float*    logitpad= (float*)   (base + carve(4));

  // 1) pad constants
  consts_kernel<<<1, 256, 0, stream>>>(b_span, w_p1, b_p1, w_p2, b_p2,
                                       w_cls, b_cls, projpad, logitpad);
  // 2) weight repack into B-fragment layout
  prep_weights<<<KT1 * NT1, 32, 0, stream>>>(w_span, wspan_p, D, NT1); // 72*48
  prep_weights<<<KT2 * NT1, 32, 0, stream>>>(w_p1,   wp1_p,   D, NT1); // 24*48
  prep_weights<<<KT2 * NT3, 32, 0, stream>>>(w_p2,   wp2_p,  PD, NT3); // 24*8
  // 3) gather span rows into A-fragment layout
  prep_A<<<MT * KT1, 32, 0, stream>>>(hidden, width_emb, a1_p);        // 96*72
  // 4) GEMM1: span = A @ w_span + b_span  (f32 + packed f16)
  gemm_kernel<0><<<MT * NT1, 32, 0, stream>>>(a1_p, wspan_p, b_span,
                                              span32, span_p, nullptr,
                                              KT1, NT1, KT2);
  // 5) logits
  logits_kernel<<<(2 * NVALID + 255) / 256, 256, 0, stream>>>(span32, w_cls,
                                                              b_cls, out);
  // 6) GEMM2: h = relu(span @ w_p1 + b_p1) (packed f16)
  gemm_kernel<1><<<MT * NT1, 32, 0, stream>>>(span_p, wp1_p, b_p1,
                                              nullptr, h_p, nullptr,
                                              KT2, NT1, KT2);
  // 7) GEMM3: proj = h @ w_p2 + b_p2 -> d_out (valid rows)
  gemm_kernel<2><<<MT * NT3, 32, 0, stream>>>(h_p, wp2_p, b_p2,
                                              nullptr, nullptr, out,
                                              KT2, NT3, 0);
  // 8) pad fill (bandwidth-dominant: ~33 MB of constant writes)
  fillpad_kernel<<<4096, 256, 0, stream>>>(projpad, logitpad, out);
}
